// Attention_4982162063852
// MI455X (gfx1250) — matrix operand order repaired and checked by reading, not compiled, not run on hardware
//
#include <hip/hip_runtime.h>

typedef __bf16 bf16_t;
typedef unsigned int u32;
typedef __attribute__((ext_vector_type(8)))  float   v8f;
typedef __attribute__((ext_vector_type(8)))  __bf16  v8bf;
typedef __attribute__((ext_vector_type(16))) __bf16  v16bf;
typedef __attribute__((ext_vector_type(4)))  u32     u32x4;
typedef __attribute__((ext_vector_type(8)))  u32     u32x8;

static constexpr int BB = 8;     // batch
static constexpr int NN = 1024;  // seq
static constexpr int CC = 1024;  // channels
static constexpr int HH = 16;    // heads
static constexpr int DD = 64;    // head dim
static constexpr float SCALE = 0.125f;  // D^-0.5

// output region offsets (elements), return order: out, attn1, q, k
static constexpr size_t OFF_ATTN1 = (size_t)BB * NN * CC;
static constexpr size_t OFF_Q     = OFF_ATTN1 + (size_t)BB * HH * NN * NN;
static constexpr size_t OFF_K     = OFF_Q + (size_t)BB * HH * NN * DD;

__device__ __forceinline__ v8f wmma_bf16(v16bf a, v16bf b, v8f c) {
  return __builtin_amdgcn_wmma_f32_16x16x32_bf16(false, a, false, b, (short)0, c,
                                                 false, false);
}

// A fragment (16x32 bf16, M x K): lane = (m | kg<<4); elem e -> K=k0+16*kg+e.
// rowptr points at row m, K contiguous.
__device__ __forceinline__ v16bf frag_a(const bf16_t* rowptr, int k0, int kg) {
  v8bf lo = *(const v8bf*)(rowptr + k0 + 16 * kg);
  v8bf hi = *(const v8bf*)(rowptr + k0 + 16 * kg + 8);
  return __builtin_shufflevector(lo, hi, 0, 1, 2, 3, 4, 5, 6, 7,
                                 8, 9, 10, 11, 12, 13, 14, 15);
}

// B fragment (32x16 bf16, K x N): lane = (n | kg<<4); elem e -> K = k0+16*kg+e.
__device__ __forceinline__ v16bf frag_b(const bf16_t* colptr, int k0, int kg) {
  return *(const v16bf*)(colptr + k0 + 16 * kg);
}

// ---- Tensor Data Mover: 2-D tile (32 cols x 64 rows, bf16) global -> LDS ----
// LDS padding: 16 DWORDs (64B row) then 4 DWORDs pad -> 40-half row pitch.
__device__ __forceinline__ void tdm_load_a_tile(u32 lds_off, const bf16_t* gsrc,
                                                u32 rows_total) {
  unsigned long long ga = (unsigned long long)(uintptr_t)gsrc;
  u32x4 g0 = {
      0x1u,                                   // count=1, user mode, no gather
      lds_off,                                // lds_addr (bytes)
      (u32)(ga & 0xffffffffu),                // global_addr[31:0]
      (u32)((ga >> 32) & 0x01ffffffu) | (2u << 30)  // addr[56:32] | type=2
  };
  u32x8 g1 = {
      (1u << 16) |                            // data_size = 2 bytes
      (1u << 20) |                            // pad_enable
      (3u << 22) |                            // pad_interval: 16 DWORDs
      (3u << 25),                             // pad_amount:   4 DWORDs
      (u32)( (CC & 0xffffu) << 16 ),          // [47:32]=0 | tensor_dim0 lo
      (u32)( (CC >> 16) | ((rows_total & 0xffffu) << 16) ),  // dim0 hi|dim1 lo
      (u32)( (rows_total >> 16) | (32u << 16) ),             // dim1 hi|tile_dim0
      64u,                                    // tile_dim1=64, tile_dim2=0
      (u32)CC,                                // tensor_dim0_stride lo
      0u,                                     // stride0 hi | dim1_stride lo
      0u                                      // dim1_stride hi
  };
  asm volatile("tensor_load_to_lds %0, %1" :: "s"(g0), "s"(g1) : "memory");
}

// ---------------- stage 0: f32 -> bf16 conversion ----------------
__global__ void cvt_f32_bf16(const float* __restrict__ src,
                             bf16_t* __restrict__ dst, int n) {
  int i = blockIdx.x * blockDim.x + threadIdx.x;
  int stride = gridDim.x * blockDim.x;
  for (; i < n; i += stride) dst[i] = (bf16_t)src[i];
}

// ---------------- stage 1: QKV GEMM (TDM-fed, 8-wave cooperative) ----------
// Block: 64 rows x 256 cols. A tile (64x32) staged by TDM, double-buffered.
// Wave: 4 m-tiles x 2 n-tiles -> 8 WMMAs per 32-K step.
__global__ __launch_bounds__(256) void qkv_gemm(
    const bf16_t* __restrict__ xb, const bf16_t* __restrict__ wb,
    float* __restrict__ q_out, float* __restrict__ k_out,
    bf16_t* __restrict__ qb, bf16_t* __restrict__ kb, bf16_t* __restrict__ vt) {
  __shared__ bf16_t abuf[2][64 * 40];  // 40-half pitch (TDM pad), 10.24 KB

  const int wave = threadIdx.x >> 5;
  const int lane = threadIdx.x & 31;
  const int kg = lane >> 4, ln = lane & 15;
  const int NBN = (3 * CC) / 256;               // 12 col-blocks
  const int mblk = blockIdx.x / NBN;            // 128 row-blocks
  const int nblk = blockIdx.x % NBN;
  const int m0 = mblk * 64;
  const int n0 = nblk * 256 + wave * 32;

  const bf16_t* agbase = xb + (size_t)m0 * CC;  // A tile origin, k advances
  const bf16_t* brow0 = wb + (size_t)(n0 + ln) * CC;
  const bf16_t* brow1 = wb + (size_t)(n0 + 16 + ln) * CC;

  const u32 lds_off[2] = { (u32)(uintptr_t)&abuf[0][0],
                           (u32)(uintptr_t)&abuf[1][0] };

  if (wave == 0) tdm_load_a_tile(lds_off[0], agbase, (u32)(BB * NN));

  v8f acc[4][2];
#pragma unroll
  for (int mt = 0; mt < 4; ++mt) { acc[mt][0] = (v8f){}; acc[mt][1] = (v8f){}; }

  const int NSTEP = CC / 32;  // 32
  for (int t = 0; t < NSTEP; ++t) {
    if (wave == 0) {
      if (t + 1 < NSTEP) {
        tdm_load_a_tile(lds_off[(t + 1) & 1], agbase + (t + 1) * 32,
                        (u32)(BB * NN));
        __builtin_amdgcn_s_wait_tensorcnt(1);  // buffer t ready
      } else {
        __builtin_amdgcn_s_wait_tensorcnt(0);
      }
    }
    __syncthreads();  // A tile t visible to all waves

    const bf16_t* at = &abuf[t & 1][0];
    const int kglob = t * 32;
    v16bf bf0 = frag_b(brow0, kglob, kg);
    v16bf bf1 = frag_b(brow1, kglob, kg);
#pragma unroll
    for (int mt = 0; mt < 4; ++mt) {
      const bf16_t* ar = at + (mt * 16 + ln) * 40;  // 80B pitch, 16B aligned
      v8bf lo = *(const v8bf*)(ar + 16 * kg);
      v8bf hi = *(const v8bf*)(ar + 16 * kg + 8);
      v16bf af = __builtin_shufflevector(lo, hi, 0, 1, 2, 3, 4, 5, 6, 7,
                                         8, 9, 10, 11, 12, 13, 14, 15);
      acc[mt][0] = wmma_bf16(af, bf0, acc[mt][0]);
      acc[mt][1] = wmma_bf16(af, bf1, acc[mt][1]);
    }
    __syncthreads();  // done reading buffer t before it is overwritten
  }

#pragma unroll
  for (int nt = 0; nt < 2; ++nt) {
    const int ncol = n0 + nt * 16;
    const int s = ncol / CC;  // 0=q, 1=k, 2=v (32-aligned, never straddles)
    const int nc = (ncol % CC) + ln;
    const int h = nc / DD, d = nc % DD;
#pragma unroll
    for (int mt = 0; mt < 4; ++mt) {
#pragma unroll
      for (int r = 0; r < 8; ++r) {
        const int m = m0 + mt * 16 + r + 8 * kg;
        const int bidx = m / NN, i = m % NN;
        const float v = acc[mt][nt][r];
        if (s == 0) {
          size_t o = ((size_t)(bidx * HH + h) * NN + i) * DD + d;
          q_out[o] = v;
          qb[o] = (bf16_t)v;
        } else if (s == 1) {
          size_t o = ((size_t)(bidx * HH + h) * NN + i) * DD + d;
          k_out[o] = v;
          kb[o] = (bf16_t)v;
        } else {
          vt[((size_t)(bidx * HH + h) * DD + d) * NN + i] = (bf16_t)v;  // V^T
        }
      }
    }
  }
}

// ---------------- stage 2: fused masked flash attention ----------------
__global__ __launch_bounds__(256) void flash_attn(
    const bf16_t* __restrict__ qb, const bf16_t* __restrict__ kb,
    const bf16_t* __restrict__ vt, const unsigned char* __restrict__ node,
    float* __restrict__ attn1, bf16_t* __restrict__ ctx) {
  __shared__ bf16_t pbuf[8][16][32];  // per-wave P staging (C/D->A relayout)

  const int wave = threadIdx.x >> 5;
  const int lane = threadIdx.x & 31;
  const int kg = lane >> 4, ln = lane & 15;
  const int task = blockIdx.x * 8 + wave;  // 8192 = B*H*(N/16)
  const int it = task & 63;
  const int bh = task >> 6;
  const int b = bh >> 4, h = bh & 15;
  const int i0 = it * 16;
  const float NEG_INF = -__builtin_inff();

  const bf16_t* qrow = qb + ((size_t)bh * NN + i0 + ln) * DD;
  const v16bf aq0 = frag_a(qrow, 0, kg);
  const v16bf aq1 = frag_a(qrow, 32, kg);

  unsigned char ni[8];
#pragma unroll
  for (int r = 0; r < 8; ++r) ni[r] = node[b * NN + i0 + r + 8 * kg];

  float mrun[8], srun[8];
  v8f o0 = {}, o1 = {}, o2 = {}, o3 = {};
#pragma unroll
  for (int r = 0; r < 8; ++r) { mrun[r] = NEG_INF; srun[r] = 0.f; }

  for (int jb = 0; jb < NN; jb += 32) {
    const bf16_t* krow0 = kb + ((size_t)bh * NN + jb + ln) * DD;
    const bf16_t* krow1 = krow0 + (size_t)16 * DD;
    v8f s0 = {}, s1 = {};
    s0 = wmma_bf16(aq0, frag_b(krow0, 0, kg), s0);
    s0 = wmma_bf16(aq1, frag_b(krow0, 32, kg), s0);
    s1 = wmma_bf16(aq0, frag_b(krow1, 0, kg), s1);
    s1 = wmma_bf16(aq1, frag_b(krow1, 32, kg), s1);

    const unsigned char nj0 = node[b * NN + jb + ln];
    const unsigned char nj1 = node[b * NN + jb + 16 + ln];

    float alpha[8];
#pragma unroll
    for (int r = 0; r < 8; ++r) {
      const float v0 = s0[r] * SCALE;
      const float v1 = s1[r] * SCALE;
      const bool mk0 = (ni[r] != nj0);
      const bool mk1 = (ni[r] != nj1);
      const float a0 = mk0 ? NEG_INF : v0;
      const float a1 = mk1 ? NEG_INF : v1;
      const int i = i0 + r + 8 * kg;
      float* arow = attn1 + ((size_t)bh * NN + i) * NN + jb;
      arow[ln] = a0;
      arow[16 + ln] = a1;

      float t = fmaxf(a0, a1);
#pragma unroll
      for (int off = 1; off < 16; off <<= 1) t = fmaxf(t, __shfl_xor(t, off, 32));
      const float nm = fmaxf(mrun[r], t);
      const float al = (nm == NEG_INF) ? 1.f : __expf(mrun[r] - nm);
      const float e0 = mk0 ? 0.f : __expf(v0 - nm);
      const float e1 = mk1 ? 0.f : __expf(v1 - nm);
      float rs = e0 + e1;
#pragma unroll
      for (int off = 1; off < 16; off <<= 1) rs += __shfl_xor(rs, off, 32);
      srun[r] = srun[r] * al + rs;
      mrun[r] = nm;
      alpha[r] = al;

      pbuf[wave][r + 8 * kg][ln] = (bf16_t)e0;
      pbuf[wave][r + 8 * kg][16 + ln] = (bf16_t)e1;
    }
#pragma unroll
    for (int r = 0; r < 8; ++r) {
      o0[r] *= alpha[r]; o1[r] *= alpha[r]; o2[r] *= alpha[r]; o3[r] *= alpha[r];
    }

    asm volatile("" ::: "memory");  // order LDS store -> LDS load (wave-internal)

    const bf16_t* prow = &pbuf[wave][ln][0];
    v8bf plo = *(const v8bf*)(prow + 16 * kg);
    v8bf phi = *(const v8bf*)(prow + 16 * kg + 8);
    v16bf ap = __builtin_shufflevector(plo, phi, 0, 1, 2, 3, 4, 5, 6, 7,
                                       8, 9, 10, 11, 12, 13, 14, 15);

    const bf16_t* vbase = vt + (size_t)bh * DD * NN + jb + 16 * kg;
    o0 = wmma_bf16(ap, *(const v16bf*)(vbase + (size_t)(0 * 16 + ln) * NN), o0);
    o1 = wmma_bf16(ap, *(const v16bf*)(vbase + (size_t)(1 * 16 + ln) * NN), o1);
    o2 = wmma_bf16(ap, *(const v16bf*)(vbase + (size_t)(2 * 16 + ln) * NN), o2);
    o3 = wmma_bf16(ap, *(const v16bf*)(vbase + (size_t)(3 * 16 + ln) * NN), o3);
  }

#pragma unroll
  for (int r = 0; r < 8; ++r) {
    const int i = i0 + r + 8 * kg;
    const float inv = (srun[r] > 0.f) ? (1.f / srun[r]) : 0.f;
    bf16_t* crow = ctx + ((size_t)(b * NN + i) * HH + h) * DD + ln;
    crow[0 * 16] = (bf16_t)(o0[r] * inv);
    crow[1 * 16] = (bf16_t)(o1[r] * inv);
    crow[2 * 16] = (bf16_t)(o2[r] * inv);
    crow[3 * 16] = (bf16_t)(o3[r] * inv);
  }
}

// ---------------- stage 3: output projection (4x2 register blocked) --------
__global__ __launch_bounds__(256) void proj_gemm(
    const bf16_t* __restrict__ ctx, const bf16_t* __restrict__ wpb,
    const float* __restrict__ bias, float* __restrict__ out) {
  const int wave = threadIdx.x >> 5;
  const int lane = threadIdx.x & 31;
  const int kg = lane >> 4, ln = lane & 15;
  const int task = blockIdx.x * 8 + wave;  // 4096 = 128 * 32
  const int tm = task / 32, tn = task % 32;
  const int m0 = tm * 64, n0 = tn * 32;

  const bf16_t* arow[4];
#pragma unroll
  for (int mt = 0; mt < 4; ++mt)
    arow[mt] = ctx + (size_t)(m0 + mt * 16 + ln) * CC;
  const bf16_t* brow0 = wpb + (size_t)(n0 + ln) * CC;
  const bf16_t* brow1 = wpb + (size_t)(n0 + 16 + ln) * CC;

  v8f acc[4][2];
#pragma unroll
  for (int mt = 0; mt < 4; ++mt) { acc[mt][0] = (v8f){}; acc[mt][1] = (v8f){}; }

#pragma unroll 2
  for (int k0 = 0; k0 < CC; k0 += 32) {
    v16bf bf0 = frag_b(brow0, k0, kg);
    v16bf bf1 = frag_b(brow1, k0, kg);
#pragma unroll
    for (int mt = 0; mt < 4; ++mt) {
      v16bf af = frag_a(arow[mt], k0, kg);
      acc[mt][0] = wmma_bf16(af, bf0, acc[mt][0]);
      acc[mt][1] = wmma_bf16(af, bf1, acc[mt][1]);
    }
  }

#pragma unroll
  for (int nt = 0; nt < 2; ++nt) {
    const float bv = bias[n0 + nt * 16 + ln];
#pragma unroll
    for (int mt = 0; mt < 4; ++mt) {
#pragma unroll
      for (int r = 0; r < 8; ++r) {
        const int m = m0 + mt * 16 + r + 8 * kg;
        out[(size_t)m * CC + n0 + nt * 16 + ln] = acc[mt][nt][r] + bv;
      }
    }
  }
}

extern "C" void kernel_launch(void* const* d_in, const int* in_sizes, int n_in,
                              void* d_out, int out_size, void* d_ws, size_t ws_size,
                              hipStream_t stream) {
  const float* x = (const float*)d_in[0];
  const unsigned char* indices = (const unsigned char*)d_in[1];  // jnp bool_
  const float* w_qkv = (const float*)d_in[2];
  const float* w_proj = (const float*)d_in[3];
  const float* b_proj = (const float*)d_in[4];

  float* out = (float*)d_out;
  float* attn1 = out + OFF_ATTN1;
  float* q_out = out + OFF_Q;
  float* k_out = out + OFF_K;

  char* ws = (char*)d_ws;
  size_t off = 0;
  auto carve = [&](size_t bytes) {
    char* p = ws + off;
    off += (bytes + 255) & ~(size_t)255;
    return p;
  };
  bf16_t* xb     = (bf16_t*)carve((size_t)BB * NN * CC * 2);
  bf16_t* wqkvb  = (bf16_t*)carve((size_t)3 * CC * CC * 2);
  bf16_t* wprojb = (bf16_t*)carve((size_t)CC * CC * 2);
  bf16_t* qb     = (bf16_t*)carve((size_t)BB * HH * NN * DD * 2);
  bf16_t* kb     = (bf16_t*)carve((size_t)BB * HH * NN * DD * 2);
  bf16_t* vt     = (bf16_t*)carve((size_t)BB * HH * DD * NN * 2);
  bf16_t* ctx    = (bf16_t*)carve((size_t)BB * NN * CC * 2);

  cvt_f32_bf16<<<4096, 256, 0, stream>>>(x, xb, BB * NN * CC);
  cvt_f32_bf16<<<4096, 256, 0, stream>>>(w_qkv, wqkvb, 3 * CC * CC);
  cvt_f32_bf16<<<2048, 256, 0, stream>>>(w_proj, wprojb, CC * CC);

  // stage 1: 128 row-blocks x 12 col-blocks
  qkv_gemm<<<1536, 256, 0, stream>>>(xb, wqkvb, q_out, k_out, qb, kb, vt);

  // stage 2: 8192 wave-tasks
  flash_attn<<<1024, 256, 0, stream>>>(qb, kb, vt, indices, attn1, ctx);

  // stage 3: 4096 wave-tasks
  proj_gemm<<<512, 256, 0, stream>>>(ctx, wprojb, b_proj, out);
}
